// LinearWaypointsPredictor_58566174048271
// MI455X (gfx1250) — matrix operand order repaired and checked by reading, not compiled, not run on hardware
//
#include <hip/hip_runtime.h>

typedef __attribute__((ext_vector_type(16))) __bf16 v16bf;
typedef __attribute__((ext_vector_type(8)))  float  v8f;
typedef __attribute__((ext_vector_type(4)))  float  v4f;

constexpr int BS   = 32768;
constexpr int NWP  = 10;
constexpr int D    = 256;
constexpr int NEXP = 6;
constexpr int H    = 64;
constexpr int OUTC = 2;

constexpr int BLOCK = 512;                 // 16 waves (wave32)
constexpr int WAVES = BLOCK / 32;
constexpr int GRID  = 256;
constexpr int TOTAL_WAVES = GRID * WAVES;  // 4096
constexpr int BPW   = BS / TOTAL_WAVES;    // 8 batches per wave

constexpr int KSTEPS = D / 32;             // 8 K-steps of 32
constexpr int NTILES = H / 16;             // 4 N-tiles of 16

struct __align__(64) Smem {
  // W1 in bf16, pre-swizzled into exact WMMA B-fragment order:
  // [expert][kstep][ntile][lane][elem]; lane<16 elems 0..7 -> K0..7, 8..15 -> K16..23; lane>=16 elems 0..7 -> K8..15, 8..15 -> K24..31
  __bf16 w1frag[NEXP][KSTEPS][NTILES][32][16];   // 196608 B
  float  rank[NWP][D + 4];                       // padded: kills LDS bank conflicts
  float  b1v[NEXP][H];
  float  w2v[NEXP][H][OUTC];
  float  b2v[NEXP][OUTC];
  float  hbuf[WAVES][16][H];                     // per-wave h scratch (f32, post-ReLU)
  float  sel[WAVES][32];                         // per-wave selected-expert outputs
};
static_assert(sizeof(Smem) <= 320 * 1024, "LDS budget");

__global__ __launch_bounds__(BLOCK)
void moe_waypoint_kernel(const float* __restrict__ x,
                         const int*   __restrict__ meas,
                         const float* __restrict__ rank_embed,
                         const float* __restrict__ W1,
                         const float* __restrict__ b1,
                         const float* __restrict__ W2,
                         const float* __restrict__ b2,
                         float*       __restrict__ out)
{
  __shared__ Smem s;
  const int tid = threadIdx.x;

  // ---- preload + swizzle weights into LDS (coalesced global reads) ----
  for (int idx = tid; idx < NEXP * D * H; idx += BLOCK) {
    const int e  = idx / (D * H);
    const int r  = idx % (D * H);
    const int k  = r / H;            // 0..255
    const int h  = r % H;            // 0..63
    const int kt = k >> 5;
    const int ko = k & 31;
    const int nt = h >> 4;
    const int ln = (h & 15) + ((ko & 8) ? 16 : 0);
    const int el = (ko & 7) + ((ko & 16) ? 8 : 0);
    s.w1frag[e][kt][nt][ln][el] = (__bf16)W1[idx];
  }
  for (int idx = tid; idx < NWP * D; idx += BLOCK) {
    s.rank[idx / D][idx % D] = rank_embed[idx];
  }
  for (int idx = tid; idx < NEXP * H; idx += BLOCK)        (&s.b1v[0][0])[idx]    = b1[idx];
  for (int idx = tid; idx < NEXP * H * OUTC; idx += BLOCK) (&s.w2v[0][0][0])[idx] = W2[idx];
  for (int idx = tid; idx < NEXP * OUTC; idx += BLOCK)     (&s.b2v[0][0])[idx]    = b2[idx];
  __syncthreads();

  const int wl    = tid >> 5;
  const int lane  = tid & 31;
  const int wave  = blockIdx.x * WAVES + wl;
  const int m     = lane & 15;      // row within 16-row M tile
  const int half  = lane >> 4;      // 0 or 1
  const bool valid = (m < NWP);

  for (int it = 0; it < BPW; ++it) {
    const int b = wave * BPW + it;
    int e = meas[b];
    e = __builtin_amdgcn_readfirstlane(e);   // uniform per wave -> SALU addressing

    // ---- build A fragments: xr = x + rank_embed, f32 add then cvt to bf16 ----
    // ISA 16-bit A 16x32 layout: lane<16 elems 0..7 -> K0..7, 8..15 -> K16..23;
    //                            lane>=16 elems 0..7 -> K8..15, 8..15 -> K24..31.
    v16bf A[KSTEPS];
    const float* xrow = x + ((size_t)b * NWP + m) * D;
#pragma unroll
    for (int kt = 0; kt < KSTEPS; ++kt) {
      const int k0 = kt * 32 + half * 8;
      float xv[16];
      if (valid) {
        v4f x0 = __builtin_nontemporal_load((const v4f*)(xrow + k0));
        v4f x1 = __builtin_nontemporal_load((const v4f*)(xrow + k0 + 4));
        v4f x2 = __builtin_nontemporal_load((const v4f*)(xrow + k0 + 16));
        v4f x3 = __builtin_nontemporal_load((const v4f*)(xrow + k0 + 20));
        const float* rr = &s.rank[m][0];
#pragma unroll
        for (int j = 0; j < 4; ++j) {
          xv[j]      = x0[j] + rr[k0 + j];
          xv[4 + j]  = x1[j] + rr[k0 + 4 + j];
          xv[8 + j]  = x2[j] + rr[k0 + 16 + j];
          xv[12 + j] = x3[j] + rr[k0 + 20 + j];
        }
      } else {
#pragma unroll
        for (int j = 0; j < 16; ++j) xv[j] = 0.f;
      }
      v16bf a;
#pragma unroll
      for (int j = 0; j < 16; ++j) a[j] = (__bf16)xv[j];
      A[kt] = a;
    }

    // ---- prefetch next batch's x rows into cache (global_prefetch_b8) ----
    // 20 valid lanes x 512B row-half x 4 lines covers the full next 10KB tile;
    // the whole GEMM1+GEMM2 below serves as latency cover.
    if (it + 1 < BPW && valid) {
      const char* p = (const char*)(x + ((size_t)(b + 1) * NWP + m) * D) + half * 512;
      __builtin_prefetch(p,       0, 0);
      __builtin_prefetch(p + 128, 0, 0);
      __builtin_prefetch(p + 256, 0, 0);
      __builtin_prefetch(p + 384, 0, 0);
    }

    // ---- GEMM1: h = relu(Xr @ W1[e] + b1[e]) via v_wmma_f32_16x16x32_bf16 ----
#pragma unroll
    for (int nt = 0; nt < NTILES; ++nt) {
      v8f acc = {0.f, 0.f, 0.f, 0.f, 0.f, 0.f, 0.f, 0.f};
#pragma unroll
      for (int kt = 0; kt < KSTEPS; ++kt) {
        const v16bf bf = *(const v16bf*)(&s.w1frag[e][kt][nt][lane][0]);  // 32B/lane contiguous
        acc = __builtin_amdgcn_wmma_f32_16x16x32_bf16(
                  false, A[kt], false, bf, (short)0, acc, false, false);
      }
      const int   col  = nt * 16 + m;           // C layout: lane%16 = N column
      const float bias = s.b1v[e][col];
#pragma unroll
      for (int r = 0; r < 8; ++r) {             // VGPR r -> M = half*8 + r
        float v = acc[r] + bias;
        s.hbuf[wl][half * 8 + r][col] = (v > 0.f) ? v : 0.f;
      }
    }

    // ---- GEMM2 (64->2) + bias, f32, selected expert; then cumsum over waypoints ----
    float sv = 0.f;
    if (lane < NWP * OUTC) {
      const int n = lane >> 1, o = lane & 1;
      float a2 = s.b2v[e][o];
#pragma unroll 16
      for (int c = 0; c < H; ++c) {
        const int cc = (c + lane) & (H - 1);    // skew -> no LDS bank conflicts
        a2 += s.hbuf[wl][n][cc] * s.w2v[e][cc][o];
      }
      sv = a2;
    }
    s.sel[wl][lane] = sv;                        // wave-private, in-order DS => no barrier
    if (lane < NWP * OUTC) {
      const int n = lane >> 1, o = lane & 1;
      float csum = 0.f;
#pragma unroll
      for (int j = 0; j < NWP; ++j) {
        if (j <= n) csum += s.sel[wl][2 * j + o];
      }
      out[(size_t)b * (NWP * OUTC) + lane] = csum;   // lane == n*2+o matches (b,n,o) layout
    }
  }
}

extern "C" void kernel_launch(void* const* d_in, const int* in_sizes, int n_in,
                              void* d_out, int out_size, void* d_ws, size_t ws_size,
                              hipStream_t stream) {
  const float* x    = (const float*)d_in[0];
  const int*   meas = (const int*)  d_in[1];
  const float* rank = (const float*)d_in[2];
  const float* W1   = (const float*)d_in[3];
  const float* b1   = (const float*)d_in[4];
  const float* W2   = (const float*)d_in[5];
  const float* b2   = (const float*)d_in[6];
  float* out = (float*)d_out;
  hipLaunchKernelGGL(moe_waypoint_kernel, dim3(GRID), dim3(BLOCK), 0, stream,
                     x, meas, rank, W1, b1, W2, b2, out);
}
